// TRMBlock_37134287241912
// MI455X (gfx1250) — matrix operand order repaired and checked by reading, compiled, not run on hardware
//
#include <hip/hip_runtime.h>

// ---------------------------------------------------------------------------
// Transformer block for MI455X (gfx1250): bf16 WMMA + async global->LDS copies.
//   x -> rmsnorm -> {q,k,v} GEMMs -> rope -> flash-attn (WMMA QK^T & PV)
//     -> wo GEMM (+resid) -> rmsnorm -> w1/w3 GEMMs -> silu*mul -> w2 GEMM (+resid)
// ---------------------------------------------------------------------------

#define B_    2
#define S_    2048
#define D_    2048
#define H_    16
#define DH_   128
#define HID_  8192
#define MROWS (B_ * S_)   // 4096
#define EPS_  1e-6f

typedef __bf16 bf16;
typedef __attribute__((ext_vector_type(16))) __bf16 bf16x16;
typedef __attribute__((ext_vector_type(8)))  __bf16 bf16x8;
typedef __attribute__((ext_vector_type(4)))  __bf16 bf16x4;
typedef __attribute__((ext_vector_type(8)))  float  f32x8;

__device__ __forceinline__ bf16x16 cat8(bf16x8 lo, bf16x8 hi) {
  return __builtin_shufflevector(lo, hi, 0,1,2,3,4,5,6,7,8,9,10,11,12,13,14,15);
}

__device__ __forceinline__ f32x8 wmma_bf16(bf16x16 a, bf16x16 b, f32x8 c) {
  return __builtin_amdgcn_wmma_f32_16x16x32_bf16(
      /*neg_a=*/false, a, /*neg_b=*/false, b,
      /*c_mod=*/(short)0, c, /*reuse_a=*/false, /*reuse_b=*/false);
}

// Async global->LDS copy, 16 bytes per lane (GVS mode: SGPR base + u32 byte
// offset per lane). Tracked by ASYNCcnt; no VGPR staging.
__device__ __forceinline__ void async_copy_b128(void* lds_dst, const void* gbase,
                                                unsigned byte_off) {
  const unsigned lds_addr = (unsigned)(uintptr_t)lds_dst;
  asm volatile("global_load_async_to_lds_b128 %0, %1, %2"
               :
               : "v"(lds_addr), "v"(byte_off), "s"(gbase)
               : "memory");
}

__device__ __forceinline__ void wait_async0() {
  asm volatile("s_wait_asynccnt 0x0" ::: "memory");
}

// ---------------------------------------------------------------------------
// Elementwise kernels
// ---------------------------------------------------------------------------

__global__ void cast_f32_bf16_kernel(const float* __restrict__ in,
                                     bf16* __restrict__ out, int n4) {
  int i = blockIdx.x * blockDim.x + threadIdx.x;
  if (i >= n4) return;
  const float4 v = ((const float4*)in)[i];
  bf16x4 o;
  o[0] = (bf16)v.x; o[1] = (bf16)v.y; o[2] = (bf16)v.z; o[3] = (bf16)v.w;
  ((bf16x4*)out)[i] = o;
}

// One block (256 thr) per row of length D_=2048.
__global__ __launch_bounds__(256) void rmsnorm_kernel(const float* __restrict__ x,
                                                      const float* __restrict__ g,
                                                      bf16* __restrict__ out) {
  const int row = blockIdx.x;
  const int t = threadIdx.x;
  const float* xr = x + (size_t)row * D_;
  float v[8];
  float ss = 0.f;
#pragma unroll
  for (int i = 0; i < 8; ++i) { v[i] = xr[t + i * 256]; ss += v[i] * v[i]; }
#pragma unroll
  for (int m = 16; m >= 1; m >>= 1) ss += __shfl_xor(ss, m, 32);
  __shared__ float red[8];
  if ((t & 31) == 0) red[t >> 5] = ss;
  __syncthreads();
  float tot = 0.f;
#pragma unroll
  for (int w = 0; w < 8; ++w) tot += red[w];
  const float rms = rsqrtf(tot * (1.0f / (float)D_) + EPS_);
  bf16* orow = out + (size_t)row * D_;
#pragma unroll
  for (int i = 0; i < 8; ++i) {
    const int c = t + i * 256;
    orow[c] = (bf16)(v[i] * rms * g[c]);
  }
}

// RoPE on fp32 q,k -> bf16; 1/sqrt(DH) folded into q. One thread per (b,s,h,d<64).
__global__ void rope_kernel(const float* __restrict__ qf, const float* __restrict__ kf,
                            bf16* __restrict__ qb, bf16* __restrict__ kb) {
  const int idx = blockIdx.x * blockDim.x + threadIdx.x;   // B*S*H*64 = 2^22
  const int d = idx & 63;
  const int h = (idx >> 6) & (H_ - 1);
  const int s = (idx >> 10) & (S_ - 1);
  const int b = idx >> 21;
  const size_t base = (((size_t)(b * S_ + s)) * H_ + h) * DH_;
  // inv_freq = 10000^(-d/64) = exp(-d * ln(10000)/64)
  const float inv = __expf((float)d * -0.14391156855f);
  const float ang = (float)s * inv;
  float sn, cs;
  __sincosf(ang, &sn, &cs);
  const float q1 = qf[base + d], q2 = qf[base + 64 + d];
  const float k1 = kf[base + d], k2 = kf[base + 64 + d];
  const float scale = 0.08838834764831845f;  // 1/sqrt(128)
  qb[base + d]      = (bf16)((q1 * cs - q2 * sn) * scale);
  qb[base + 64 + d] = (bf16)((q2 * cs + q1 * sn) * scale);
  kb[base + d]      = (bf16)(k1 * cs - k2 * sn);
  kb[base + 64 + d] = (bf16)(k2 * cs + k1 * sn);
}

__global__ void silu_mul_bf16_kernel(const bf16* __restrict__ u1,
                                     const bf16* __restrict__ u3,
                                     bf16* __restrict__ out, int n4) {
  int i = blockIdx.x * blockDim.x + threadIdx.x;
  if (i >= n4) return;
  const bf16x4 a = ((const bf16x4*)u1)[i];
  const bf16x4 c = ((const bf16x4*)u3)[i];
  bf16x4 o;
#pragma unroll
  for (int e = 0; e < 4; ++e) {
    const float av = (float)a[e];
    const float cv = (float)c[e];
    o[e] = (bf16)(av / (1.f + __expf(-av)) * cv);
  }
  ((bf16x4*)out)[i] = o;
}

// ---------------------------------------------------------------------------
// bf16 WMMA GEMM: C[M,N] = A[M,K] @ B[K,N] (+resid), A/B row-major bf16.
// 256 threads = 8 waves; block tile 128x128x32; wave tile 32x64 (2x4 WMMA).
// A tile streamed via async global->LDS; B tile transposed through VGPRs.
// ---------------------------------------------------------------------------

#define BM 128
#define BN 128
#define BK 32
#define LPAD (BK + 8)   // 40 elems = 80B row stride: 16B aligned, conflict-free

template <bool RESID, bool OUTBF>
__global__ __launch_bounds__(256) void gemm_bf16_kernel(
    const bf16* __restrict__ A, const bf16* __restrict__ Bm,
    const float* __restrict__ resid, float* __restrict__ Cf,
    bf16* __restrict__ Cb, int M, int N, int K) {
  __shared__ __align__(16) bf16 sA[2][BM][LPAD];
  __shared__ __align__(16) bf16 sB[2][BN][LPAD];  // stored transposed: [n][k]

  const int t = threadIdx.x;
  const int lane = t & 31, wid = t >> 5;
  const int half = lane >> 4, l16 = lane & 15;
  const int wm = (wid & 3) * 32, wn = (wid >> 2) * 64;
  const int m0 = blockIdx.y * BM, n0 = blockIdx.x * BN;

  const int arow = t >> 1, aseg = (t & 1) * 16;   // 128 rows x 32 cols
  const int brow = t >> 3, bseg = (t & 7) * 16;   // 32 rows x 128 cols
  const int ntiles = K / BK;

  auto load_tile = [&](int kt, int buf) {
    // A tile: two async 16B copies per thread, no VGPR staging
    const unsigned abyte =
        (unsigned)(((size_t)(m0 + arow) * K + (size_t)kt * BK + aseg) * 2);
    async_copy_b128(&sA[buf][arow][aseg], A, abyte);
    async_copy_b128(&sA[buf][arow][aseg + 8], A, abyte + 16u);
    // B tile: load 16 elems along N, scatter transposed into LDS
    const bf16* bp = Bm + (size_t)(kt * BK + brow) * N + n0 + bseg;
    const bf16x8 b0 = *(const bf16x8*)(bp);
    const bf16x8 b1 = *(const bf16x8*)(bp + 8);
#pragma unroll
    for (int i = 0; i < 8; ++i) {
      sB[buf][bseg + i][brow]     = b0[i];
      sB[buf][bseg + 8 + i][brow] = b1[i];
    }
    if (kt + 2 < ntiles) {  // global_prefetch_b8 for the tile after next
      __builtin_prefetch(Bm + (size_t)((kt + 2) * BK + brow) * N + n0 + bseg, 0, 1);
    }
  };

  const f32x8 zero8 = {0.f, 0.f, 0.f, 0.f, 0.f, 0.f, 0.f, 0.f};
  f32x8 acc[2][4];
#pragma unroll
  for (int f = 0; f < 2; ++f)
#pragma unroll
    for (int j = 0; j < 4; ++j) acc[f][j] = zero8;

  load_tile(0, 0);
  for (int kt = 0; kt < ntiles; ++kt) {
    wait_async0();     // async A tile (this wave's part) has landed in LDS
    __syncthreads();   // everyone's A/B tile parts visible
    if (kt + 1 < ntiles) load_tile(kt + 1, (kt + 1) & 1);
    const int buf = kt & 1;
    bf16x16 af[2];
#pragma unroll
    for (int f = 0; f < 2; ++f) {
      const bf16* ar = &sA[buf][wm + f * 16 + l16][0];
      af[f] = cat8(*(const bf16x8*)(ar + half * 16),
                   *(const bf16x8*)(ar + half * 16 + 8));
    }
#pragma unroll
    for (int j = 0; j < 4; ++j) {
      const bf16x16 bfr = *(const bf16x16*)&sB[buf][wn + j * 16 + l16][half * 16];
#pragma unroll
      for (int f = 0; f < 2; ++f) acc[f][j] = wmma_bf16(af[f], bfr, acc[f][j]);
    }
    __syncthreads();   // compute done before next iter's tile overwrites
  }

  // epilogue: C layout row = r + 8*half, col = lane&15 (per n-tile)
#pragma unroll
  for (int f = 0; f < 2; ++f) {
#pragma unroll
    for (int r = 0; r < 8; ++r) {
      const int row = m0 + wm + f * 16 + r + 8 * half;
#pragma unroll
      for (int j = 0; j < 4; ++j) {
        const int col = n0 + wn + j * 16 + l16;
        float v = acc[f][j][r];
        if (RESID) v += resid[(size_t)row * N + col];
        if (OUTBF) Cb[(size_t)row * N + col] = (bf16)v;
        else       Cf[(size_t)row * N + col] = v;
      }
    }
  }
}

// ---------------------------------------------------------------------------
// Flash attention (non-causal): per block one (b,h,64-row q tile); 4 waves.
// Q*K^T and P*V via WMMA bf16; Q/K tiles streamed with async global->LDS.
// ---------------------------------------------------------------------------

__global__ __launch_bounds__(128) void flash_attn_kernel(
    const bf16* __restrict__ q, const bf16* __restrict__ k,
    const bf16* __restrict__ v, bf16* __restrict__ o) {
  __shared__ __align__(16) bf16 sQ[64][DH_ + 8];   // 272B rows
  __shared__ __align__(16) bf16 sK[64][DH_ + 8];   // row-major K == transposed B
  __shared__ __align__(16) bf16 sVt[DH_][64 + 8];  // V transposed: [d][s]
  __shared__ __align__(16) bf16 sP[64][64 + 8];    // probabilities, row-major

  const int t = threadIdx.x;
  const int lane = t & 31, wid = t >> 5;
  const int half = lane >> 4, l16 = lane & 15;
  const int qt = blockIdx.x, h = blockIdx.y, b = blockIdx.z;

  // issue async Q tile copy (scale already folded in at rope time)
#pragma unroll
  for (int i = 0; i < 8; ++i) {
    const int sidx = t * 8 + i;
    const int row = sidx >> 4, seg = (sidx & 15) * 8;
    const size_t g = (((size_t)(b * S_ + qt * 64 + row)) * H_ + h) * DH_ + seg;
    async_copy_b128(&sQ[row][seg], q, (unsigned)(g * 2));
  }

  const f32x8 zero8 = {0.f, 0.f, 0.f, 0.f, 0.f, 0.f, 0.f, 0.f};
  f32x8 oa[8];
  float mr[8], lr[8];
#pragma unroll
  for (int nt = 0; nt < 8; ++nt) oa[nt] = zero8;
#pragma unroll
  for (int r = 0; r < 8; ++r) { mr[r] = -3.0e38f; lr[r] = 0.f; }

  for (int kt = 0; kt < S_ / 64; ++kt) {
    __syncthreads();  // previous iteration's reads of sK/sVt are done
#pragma unroll
    for (int i = 0; i < 8; ++i) {
      const int sidx = t * 8 + i;
      const int row = sidx >> 4, seg = (sidx & 15) * 8;
      const size_t g = (((size_t)(b * S_ + kt * 64 + row)) * H_ + h) * DH_ + seg;
      async_copy_b128(&sK[row][seg], k, (unsigned)(g * 2));     // K: async
      const bf16x8 vv = *(const bf16x8*)(v + g);                // V: transpose
#pragma unroll
      for (int e = 0; e < 8; ++e) sVt[seg + e][row] = vv[e];
    }
    wait_async0();    // K tile (+ Q on first iteration) landed
    __syncthreads();

    // --- scores: wave's 16-row strip x 64 keys ---
    f32x8 sc[4];
#pragma unroll
    for (int j = 0; j < 4; ++j) sc[j] = zero8;
#pragma unroll
    for (int c = 0; c < 4; ++c) {  // DH/32 k-chunks
      const bf16* ar = &sQ[wid * 16 + l16][c * 32];
      const bf16x16 af = cat8(*(const bf16x8*)(ar + half * 16),
                              *(const bf16x8*)(ar + half * 16 + 8));
#pragma unroll
      for (int j = 0; j < 4; ++j) {
        const bf16x16 bfr = *(const bf16x16*)&sK[j * 16 + l16][c * 32 + half * 16];
        sc[j] = wmma_bf16(af, bfr, sc[j]);
      }
    }

    // --- online softmax (row = wid*16 + r + 8*half) ---
#pragma unroll
    for (int r = 0; r < 8; ++r) {
      float mx = fmaxf(fmaxf(sc[0][r], sc[1][r]), fmaxf(sc[2][r], sc[3][r]));
      mx = fmaxf(mx, __shfl_xor(mx, 1, 32));
      mx = fmaxf(mx, __shfl_xor(mx, 2, 32));
      mx = fmaxf(mx, __shfl_xor(mx, 4, 32));
      mx = fmaxf(mx, __shfl_xor(mx, 8, 32));
      const float mn = fmaxf(mr[r], mx);
      const float alpha = __expf(mr[r] - mn);
      const int prow = wid * 16 + r + 8 * half;
      float sum = 0.f;
#pragma unroll
      for (int j = 0; j < 4; ++j) {
        const float p = __expf(sc[j][r] - mn);
        sum += p;
        sP[prow][j * 16 + l16] = (bf16)p;
      }
      sum += __shfl_xor(sum, 1, 32);
      sum += __shfl_xor(sum, 2, 32);
      sum += __shfl_xor(sum, 4, 32);
      sum += __shfl_xor(sum, 8, 32);
      lr[r] = lr[r] * alpha + sum;
      mr[r] = mn;
#pragma unroll
      for (int nt = 0; nt < 8; ++nt) oa[nt][r] = oa[nt][r] * alpha;
    }

    // --- P @ V: A = P strip [16x64], B = V [64x128] ---
#pragma unroll
    for (int c2 = 0; c2 < 2; ++c2) {
      const bf16* ar = &sP[wid * 16 + l16][c2 * 32];
      const bf16x16 af = cat8(*(const bf16x8*)(ar + half * 16),
                              *(const bf16x8*)(ar + half * 16 + 8));
#pragma unroll
      for (int nt = 0; nt < 8; ++nt) {
        const bf16x16 bfr = *(const bf16x16*)&sVt[nt * 16 + l16][c2 * 32 + half * 16];
        oa[nt] = wmma_bf16(af, bfr, oa[nt]);
      }
    }
  }

  // write ctx (bf16, layout [B,S,H,DH])
#pragma unroll
  for (int r = 0; r < 8; ++r) {
    const float inv = 1.f / lr[r];
    const int row = qt * 64 + wid * 16 + r + 8 * half;
    const size_t base = (((size_t)(b * S_ + row)) * H_ + h) * DH_;
#pragma unroll
    for (int nt = 0; nt < 8; ++nt)
      o[base + nt * 16 + l16] = (bf16)(oa[nt][r] * inv);
  }
}

// ---------------------------------------------------------------------------
// Host orchestration
// ---------------------------------------------------------------------------

extern "C" void kernel_launch(void* const* d_in, const int* in_sizes, int n_in,
                              void* d_out, int out_size, void* d_ws, size_t ws_size,
                              hipStream_t stream) {
  (void)in_sizes; (void)n_in; (void)out_size; (void)ws_size;
  const float* x  = (const float*)d_in[0];
  const float* wq = (const float*)d_in[1];
  const float* wk = (const float*)d_in[2];
  const float* wv = (const float*)d_in[3];
  const float* wo = (const float*)d_in[4];
  const float* g1 = (const float*)d_in[5];
  const float* g2 = (const float*)d_in[6];
  const float* w1 = (const float*)d_in[7];
  const float* w2 = (const float*)d_in[8];
  const float* w3 = (const float*)d_in[9];

  const size_t DD   = (size_t)D_ * D_;        // 4.19M
  const size_t DHID = (size_t)D_ * HID_;      // 16.8M
  const size_t ND   = (size_t)MROWS * D_;     // 8.39M
  const size_t NH   = (size_t)MROWS * HID_;   // 33.6M

  char* ws = (char*)d_ws;
  size_t off = 0;
  auto alloc = [&](size_t bytes) -> void* {
    void* p = ws + off;
    off += (bytes + 255) & ~(size_t)255;
    return p;
  };

  bf16* wq_b  = (bf16*)alloc(DD * 2);
  bf16* wk_b  = (bf16*)alloc(DD * 2);
  bf16* wv_b  = (bf16*)alloc(DD * 2);
  bf16* wo_b  = (bf16*)alloc(DD * 2);
  bf16* w1_b  = (bf16*)alloc(DHID * 2);
  bf16* w2_b  = (bf16*)alloc(DHID * 2);
  bf16* w3_b  = (bf16*)alloc(DHID * 2);
  bf16* h_b   = (bf16*)alloc(ND * 2);
  float* qf   = (float*)alloc(ND * 4);
  float* kf   = (float*)alloc(ND * 4);
  float* vf   = (float*)alloc(ND * 4);
  bf16* q_b   = (bf16*)alloc(ND * 2);
  bf16* k_b   = (bf16*)alloc(ND * 2);
  bf16* v_b   = (bf16*)alloc(ND * 2);
  bf16* ctx_b = (bf16*)alloc(ND * 2);
  float* x1   = (float*)alloc(ND * 4);
  bf16* h2_b  = (bf16*)alloc(ND * 2);
  bf16* u1_b  = (bf16*)alloc(NH * 2);
  bf16* u3_b  = (bf16*)alloc(NH * 2);
  bf16* g_b   = (bf16*)alloc(NH * 2);

  // weight casts fp32 -> bf16
  cast_f32_bf16_kernel<<<(int)(DD / 4 / 256), 256, 0, stream>>>(wq, wq_b, (int)(DD / 4));
  cast_f32_bf16_kernel<<<(int)(DD / 4 / 256), 256, 0, stream>>>(wk, wk_b, (int)(DD / 4));
  cast_f32_bf16_kernel<<<(int)(DD / 4 / 256), 256, 0, stream>>>(wv, wv_b, (int)(DD / 4));
  cast_f32_bf16_kernel<<<(int)(DD / 4 / 256), 256, 0, stream>>>(wo, wo_b, (int)(DD / 4));
  cast_f32_bf16_kernel<<<(int)(DHID / 4 / 256), 256, 0, stream>>>(w1, w1_b, (int)(DHID / 4));
  cast_f32_bf16_kernel<<<(int)(DHID / 4 / 256), 256, 0, stream>>>(w2, w2_b, (int)(DHID / 4));
  cast_f32_bf16_kernel<<<(int)(DHID / 4 / 256), 256, 0, stream>>>(w3, w3_b, (int)(DHID / 4));

  // attention branch
  rmsnorm_kernel<<<MROWS, 256, 0, stream>>>(x, g1, h_b);
  const dim3 gD(D_ / BN, MROWS / BM);      // (16, 32)
  gemm_bf16_kernel<false, false><<<gD, 256, 0, stream>>>(h_b, wq_b, nullptr, qf, nullptr, MROWS, D_, D_);
  gemm_bf16_kernel<false, false><<<gD, 256, 0, stream>>>(h_b, wk_b, nullptr, kf, nullptr, MROWS, D_, D_);
  gemm_bf16_kernel<false, false><<<gD, 256, 0, stream>>>(h_b, wv_b, nullptr, vf, nullptr, MROWS, D_, D_);
  rope_kernel<<<(B_ * S_ * H_ * 64) / 256, 256, 0, stream>>>(qf, kf, q_b, k_b);
  cast_f32_bf16_kernel<<<(int)(ND / 4 / 256), 256, 0, stream>>>(vf, v_b, (int)(ND / 4));
  flash_attn_kernel<<<dim3(S_ / 64, H_, B_), 128, 0, stream>>>(q_b, k_b, v_b, ctx_b);
  gemm_bf16_kernel<true, false><<<gD, 256, 0, stream>>>(ctx_b, wo_b, x, x1, nullptr, MROWS, D_, D_);

  // FFN branch
  rmsnorm_kernel<<<MROWS, 256, 0, stream>>>(x1, g2, h2_b);
  const dim3 gF(HID_ / BN, MROWS / BM);    // (64, 32)
  gemm_bf16_kernel<false, true><<<gF, 256, 0, stream>>>(h2_b, w1_b, nullptr, nullptr, u1_b, MROWS, HID_, D_);
  gemm_bf16_kernel<false, true><<<gF, 256, 0, stream>>>(h2_b, w3_b, nullptr, nullptr, u3_b, MROWS, HID_, D_);
  silu_mul_bf16_kernel<<<(int)(NH / 4 / 256), 256, 0, stream>>>(u1_b, u3_b, g_b, (int)(NH / 4));
  gemm_bf16_kernel<true, false><<<gD, 256, 0, stream>>>(g_b, w2_b, x1, (float*)d_out, nullptr, MROWS, D_, HID_);
}
